// GNNAutoencoder_30597347017287
// MI455X (gfx1250) — matrix operand order repaired and checked by reading, not compiled, not run on hardware
//
#include <hip/hip_runtime.h>
#include <hip/hip_bf16.h>

// GNN autoencoder forward on MI455X (gfx1250, wave32).
// Big GEMMs: v_wmma_f32_16x16x32_bf16, double-buffered LDS software pipeline.
// Weights are pre-converted to bf16 and pre-transposed to [N][K] once, so the
// B tile is staged by the Tensor Data Mover (tensor_load_to_lds, TENSORcnt),
// with TDM padding reproducing the bank-conflict-free 80B LDS row stride.
// A tiles (activations) are staged through registers with v_cvt_pk_bf16_f32.

#define N_NODES 8192
#define IN_DIM  20000
#define HID     256
#define LAT     64
#define N_EDGES 262144

typedef __attribute__((ext_vector_type(16))) __bf16 v16bf;
typedef __attribute__((ext_vector_type(8)))  __bf16 v8bf;
typedef __attribute__((ext_vector_type(2)))  __bf16 v2bf;
typedef __attribute__((ext_vector_type(8)))  float  v8f;
typedef __attribute__((ext_vector_type(4)))  unsigned v4u;
typedef __attribute__((ext_vector_type(8)))  int      v8i;
typedef __attribute__((ext_vector_type(4)))  int      v4i;

__device__ __forceinline__ unsigned pack2_bf16(float x, float y) {
    v2bf v; v[0] = (__bf16)x; v[1] = (__bf16)y;   // -> v_cvt_pk_bf16_f32
    return __builtin_bit_cast(unsigned, v);
}

// 32B fragment load as two 16B LDS loads (80B row stride keeps 16B alignment).
__device__ __forceinline__ v16bf load_frag(const __bf16* p) {
    v8bf lo = *(const v8bf*)p;
    v8bf hi = *(const v8bf*)(p + 8);
    return __builtin_shufflevector(lo, hi, 0, 1, 2, 3, 4, 5, 6, 7,
                                           8, 9, 10, 11, 12, 13, 14, 15);
}

// A-fragment K order: per lane-half h the fragment order is
// [h*16+0..15] (contiguous, matches B); LDS store position for K value k:
__device__ __forceinline__ int a_pos(int k) {
    return k;
}

// ---------------------------------------------------------------------------
// TDM: DMA one B tile (tile_dim0=32 K-elems x tile_dim1=128 N-rows, bf16) from
// the pre-transposed weight matrix Bt[N][K] into LDS. Row payload = 64B; TDM
// padding (pad_interval=16 DWORDs, pad_amount=4 DWORDs) yields an 80B LDS row
// stride == SBS*2. Rows beyond rem_n read OOB -> zeros. D# per ISA 8.3/8.4.
// This toolchain exposes the 6-arg builtin (extra int32x8 group, then cpol).
// ---------------------------------------------------------------------------
__device__ __forceinline__ void tdm_load_b_tile(const __bf16* gsrc, unsigned lds_addr,
                                                unsigned rem_k, unsigned rem_n,
                                                unsigned kstride /*elements*/) {
    unsigned long long ga = (unsigned long long)(size_t)gsrc;
    v4u g0;
    g0[0] = 1u;                                           // count=1 (valid user D#)
    g0[1] = lds_addr;                                     // LDS byte address
    g0[2] = (unsigned)(ga & 0xFFFFFFFFu);                 // global_addr[31:0]
    g0[3] = (unsigned)((ga >> 32) & 0x01FFFFFFu) | (2u << 30); // [56:32] | type=2
    const unsigned w0 = (1u << 16)   // data_size = 2 bytes
                      | (1u << 20)   // pad_enable (load only)
                      | (3u << 22)   // pad_interval: 16 DWORDs (64B)
                      | (3u << 25);  // pad_amount:   4 DWORDs (16B)
    v8i g1;
    g1[0] = (int)w0;
    g1[1] = (int)((rem_k & 0xFFFFu) << 16);               // tensor_dim0[15:0]
    g1[2] = (int)(((rem_k >> 16) & 0xFFFFu) | ((rem_n & 0xFFFFu) << 16));
    g1[3] = (int)(((rem_n >> 16) & 0xFFFFu) | (32u << 16));   // tile_dim0 = 32
    g1[4] = (int)128u;                                    // tile_dim1=128, tile_dim2=0
    g1[5] = (int)kstride;                                 // tensor_dim0_stride[31:0]
    g1[6] = 0;                                            // stride[47:32], dim1_stride lo
    g1[7] = 0;
    v4i z4 = {0, 0, 0, 0};                                // 2D: groups 2/3 unused
    v8i z8 = {0, 0, 0, 0, 0, 0, 0, 0};
    __builtin_amdgcn_tensor_load_to_lds(g0, g1, z4, z4, z8, 0);
}

// ---------------------------------------------------------------------------
// Tiled WMMA GEMM: C[M,N] = act(A[M,K] @ Bt^T + bias), Bt is bf16 [N][K].
// Block = 256 threads = 8 waves; block tile 128x128, wave tile 32x64
// (2 A-frags x 4 B-frags = 8 wmma/K-step). Double-buffered LDS: A-tile global
// loads for step k+1 issue before the WMMAs of step k; B tile k+1 is DMA'd by
// the TDM concurrently and joined with s_wait_tensorcnt before the barrier.
// Requires M % 128 == 0, K % 32 == 0 (true here); N bounds-checked.
// ---------------------------------------------------------------------------
template <bool RELU_A, bool RELU_OUT, bool NT_OUT>
__global__ void __launch_bounds__(256)
gemm_bf16_wmma(const float* __restrict__ A, const __bf16* __restrict__ Bt,
               const float* __restrict__ bias, float* __restrict__ C,
               int M, int N, int K)
{
    constexpr int BM = 128, BN = 128, BK = 32;
    constexpr int SAS = 40, SBS = 40;          // padded LDS row strides (bf16)
    constexpr int ABUF = BM * SAS, BBUF = BN * SBS;
    __shared__ __bf16 sAf[2 * ABUF];           // ping-pong A tiles (frag order)
    __shared__ __bf16 sBf[2 * BBUF];           // ping-pong B tiles [n][k]

    const int tid  = threadIdx.x;
    const int wave = tid >> 5;
    const int lane = tid & 31;
    const int r    = lane & 15;
    const int h    = lane >> 4;
    const int m0   = blockIdx.x * BM;
    const int n0   = blockIdx.y * BN;
    const int wm0  = (wave & 3) * 32;
    const int wn0  = (wave >> 2) * 64;

    // ---- per-thread A staging geometry ----
    const int rA  = tid >> 3;                  // 0..31
    const int kqA = (tid & 7) << 2;            // 0,4,..,28
    const float* aptr = A + (size_t)(m0 + rA) * K + kqA;
    const size_t aRowStride = (size_t)32 * K;
    const int    aDst = rA * SAS + a_pos(kqA);

    // ---- TDM source geometry (uniform) ----
    const __bf16* btile = Bt + (size_t)n0 * K;       // + k0 per step
    const unsigned rem_n = (unsigned)((n0 < N) ? (N - n0) : 0);
    const unsigned ldsB0 = (unsigned)(size_t)&sBf[0];

    const v8f vzero = {0.f, 0.f, 0.f, 0.f, 0.f, 0.f, 0.f, 0.f};
    v8f acc0[4], acc1[4];
    #pragma unroll
    for (int t = 0; t < 4; ++t) { acc0[t] = vzero; acc1[t] = vzero; }

    float a_reg[16];

    auto g_loadA = [&]() {
        #pragma unroll
        for (int i = 0; i < 4; ++i) {
            float4 f = *(const float4*)(aptr + (size_t)i * aRowStride);
            a_reg[4 * i + 0] = f.x; a_reg[4 * i + 1] = f.y;
            a_reg[4 * i + 2] = f.z; a_reg[4 * i + 3] = f.w;
        }
        __builtin_prefetch(aptr + 2 * BK, 0, 1);   // A stream, 2 tiles ahead
    };

    auto l_storeA = [&](int buf) {
        __bf16* sa = sAf + buf * ABUF;
        #pragma unroll
        for (int i = 0; i < 4; ++i) {
            float x0 = a_reg[4 * i + 0], x1 = a_reg[4 * i + 1];
            float x2 = a_reg[4 * i + 2], x3 = a_reg[4 * i + 3];
            if (RELU_A) {
                x0 = fmaxf(x0, 0.f); x1 = fmaxf(x1, 0.f);
                x2 = fmaxf(x2, 0.f); x3 = fmaxf(x3, 0.f);
            }
            uint2 pk; pk.x = pack2_bf16(x0, x1); pk.y = pack2_bf16(x2, x3);
            *(uint2*)(sa + aDst + i * 32 * SAS) = pk;
        }
    };

    auto compute = [&](int buf) {
        const __bf16* sa = sAf + buf * ABUF;
        const __bf16* sb = sBf + buf * BBUF;
        v16bf a0 = load_frag(sa + (wm0 + r) * SAS + h * 16);
        v16bf a1 = load_frag(sa + (wm0 + 16 + r) * SAS + h * 16);
        v16bf bf[4];
        #pragma unroll
        for (int t = 0; t < 4; ++t)
            bf[t] = load_frag(sb + (wn0 + t * 16 + r) * SBS + h * 16);
        #pragma unroll
        for (int t = 0; t < 4; ++t) {
            acc0[t] = __builtin_amdgcn_wmma_f32_16x16x32_bf16(
                false, a0, false, bf[t], (short)0, acc0[t], false, false);
            acc1[t] = __builtin_amdgcn_wmma_f32_16x16x32_bf16(
                false, a1, false, bf[t], (short)0, acc1[t], false, false);
        }
    };

    // ---- software pipeline: prologue (tile 0) ----
    const int nk = K / BK;
    g_loadA();
    if (wave == 0)
        tdm_load_b_tile(btile, ldsB0, (unsigned)K, rem_n, (unsigned)K);
    l_storeA(0);
    aptr += BK;
    if (wave == 0) __builtin_amdgcn_s_wait_tensorcnt((short)0);
    __syncthreads();

    for (int kk = 0; kk < nk; ++kk) {
        const bool more = (kk + 1) < nk;
        if (more) {
            g_loadA();                         // A tile kk+1 -> registers
            if (wave == 0) {                   // B tile kk+1 -> LDS via TDM
                int k1 = (kk + 1) * BK;
                tdm_load_b_tile(btile + k1,
                                ldsB0 + (unsigned)(((kk + 1) & 1) * BBUF * 2),
                                (unsigned)(K - k1), rem_n, (unsigned)K);
            }
        }
        compute(kk & 1);
        if (more) {
            l_storeA((kk + 1) & 1);            // wait+cvt+store after WMMAs
            aptr += BK;
            if (wave == 0) __builtin_amdgcn_s_wait_tensorcnt((short)0);
        }
        __syncthreads();
    }

    // ---- epilogue: D layout: row = v + 8*h, col = lane%16 ----
    #pragma unroll
    for (int t = 0; t < 4; ++t) {
        int n = n0 + wn0 + t * 16 + r;
        if (n < N) {
            float bv = bias ? bias[n] : 0.f;
            #pragma unroll
            for (int v = 0; v < 8; ++v) {
                int m = m0 + wm0 + (h << 3) + v;
                float v0 = acc0[t][v] + bv;
                float v1 = acc1[t][v] + bv;
                if (RELU_OUT) { v0 = fmaxf(v0, 0.f); v1 = fmaxf(v1, 0.f); }
                float* p0 = &C[(size_t)m * N + n];
                float* p1 = &C[(size_t)(m + 16) * N + n];
                if (NT_OUT) {
                    __builtin_nontemporal_store(v0, p0);
                    __builtin_nontemporal_store(v1, p1);
                } else {
                    *p0 = v0; *p1 = v1;
                }
            }
        }
    }
}

// ---------------------------------------------------------------------------
// Weight pre-pass: W[K,N] f32 -> Wt[N,K] bf16 (32x32 LDS tiles, coalesced).
// K, N multiples of 32 for all four weights.
// ---------------------------------------------------------------------------
__global__ void __launch_bounds__(256)
transpose_f32_to_bf16(const float* __restrict__ W, __bf16* __restrict__ Wt,
                      int K, int N) {
    __shared__ __bf16 tile[32][34];
    const int n0 = blockIdx.x * 32, k0 = blockIdx.y * 32;
    const int tx = threadIdx.x & 31;
    const int ty = threadIdx.x >> 5;           // 0..7
    #pragma unroll
    for (int i = 0; i < 4; ++i)
        tile[ty + i * 8][tx] = (__bf16)W[(size_t)(k0 + ty + i * 8) * N + n0 + tx];
    __syncthreads();
    #pragma unroll
    for (int i = 0; i < 4; ++i)
        Wt[(size_t)(n0 + ty + i * 8) * K + k0 + tx] = tile[tx][ty + i * 8];
}

// ---------------------------------------------------------------------------
// GCN normalization + scatter stage
// ---------------------------------------------------------------------------
__global__ void deg_init_kernel(float* __restrict__ deg) {
    int i = blockIdx.x * blockDim.x + threadIdx.x;
    if (i < N_NODES) deg[i] = 1.0f;            // self loop
}

__global__ void deg_edges_kernel(const int* __restrict__ ei, float* __restrict__ deg) {
    int e = blockIdx.x * blockDim.x + threadIdx.x;
    if (e < N_EDGES) atomicAdd(&deg[ei[N_EDGES + e]], 1.0f);
}

__global__ void dinv_kernel(const float* __restrict__ deg, float* __restrict__ dinv) {
    int i = blockIdx.x * blockDim.x + threadIdx.x;
    if (i < N_NODES) {
        float d = deg[i];
        dinv[i] = (d > 0.f) ? rsqrtf(d) : 0.f;
    }
}

__global__ void agg_init_kernel(const float* __restrict__ h0,
                                const float* __restrict__ dinv,
                                const float* __restrict__ b_gcn,
                                float* __restrict__ hagg) {
    int i = blockIdx.x, c = threadIdx.x;
    float di = dinv[i];
    hagg[(size_t)i * HID + c] = h0[(size_t)i * HID + c] * di * di + b_gcn[c];
}

__global__ void agg_edges_kernel(const int* __restrict__ ei,
                                 const float* __restrict__ dinv,
                                 const float* __restrict__ h0,
                                 float* __restrict__ hagg) {
    int e = blockIdx.x, c = threadIdx.x;
    int row = ei[e];
    int col = ei[N_EDGES + e];
    float norm = dinv[row] * dinv[col];
    atomicAdd(&hagg[(size_t)col * HID + c], h0[(size_t)row * HID + c] * norm);
}

// ---------------------------------------------------------------------------
extern "C" void kernel_launch(void* const* d_in, const int* in_sizes, int n_in,
                              void* d_out, int out_size, void* d_ws, size_t ws_size,
                              hipStream_t stream) {
    const float* x      = (const float*)d_in[0];
    const int*   ei     = (const int*)  d_in[1];   // [2, N_EDGES] flat
    const float* w_gcn  = (const float*)d_in[2];
    const float* b_gcn  = (const float*)d_in[3];
    const float* w_enc  = (const float*)d_in[4];
    const float* b_enc  = (const float*)d_in[5];
    const float* w_d1   = (const float*)d_in[6];
    const float* b_d1   = (const float*)d_in[7];
    const float* w_d2   = (const float*)d_in[8];
    const float* b_d2   = (const float*)d_in[9];

    float* recon  = (float*)d_out;                              // [N_NODES, IN_DIM]
    float* latent = recon + (size_t)N_NODES * IN_DIM;           // [N_NODES, LAT]

    // workspace layout (~46 MB)
    float* h0   = (float*)d_ws;                                 // [N_NODES, HID]
    float* hagg = h0   + (size_t)N_NODES * HID;
    float* tbuf = hagg + (size_t)N_NODES * HID;
    float* deg  = tbuf + (size_t)N_NODES * HID;
    float* dinv = deg  + N_NODES;
    __bf16* wgcn_t = (__bf16*)(dinv + N_NODES);                 // [HID, IN_DIM]
    __bf16* wenc_t = wgcn_t + (size_t)IN_DIM * HID;             // [LAT, HID]
    __bf16* wd1_t  = wenc_t + (size_t)HID * LAT;                // [HID, LAT]
    __bf16* wd2_t  = wd1_t  + (size_t)LAT * HID;                // [IN_DIM, HID]

    // 0) weight pre-pass: f32 [K,N] -> bf16 [N,K] (amortized over M=8192 rows)
    transpose_f32_to_bf16<<<dim3(HID / 32, IN_DIM / 32), 256, 0, stream>>>(w_gcn, wgcn_t, IN_DIM, HID);
    transpose_f32_to_bf16<<<dim3(LAT / 32, HID / 32),    256, 0, stream>>>(w_enc, wenc_t, HID, LAT);
    transpose_f32_to_bf16<<<dim3(HID / 32, LAT / 32),    256, 0, stream>>>(w_d1,  wd1_t,  LAT, HID);
    transpose_f32_to_bf16<<<dim3(IN_DIM / 32, HID / 32), 256, 0, stream>>>(w_d2,  wd2_t,  HID, IN_DIM);

    // 1) h0 = x @ w_gcn                       (84 GFLOP; x streamed ~2x)
    gemm_bf16_wmma<false, false, false><<<dim3(N_NODES / 128, HID / 128), 256, 0, stream>>>(
        x, wgcn_t, nullptr, h0, N_NODES, HID, IN_DIM);

    // 2) GCN normalization + scatter
    deg_init_kernel <<<N_NODES / 256, 256, 0, stream>>>(deg);
    deg_edges_kernel<<<N_EDGES / 256, 256, 0, stream>>>(ei, deg);
    dinv_kernel     <<<N_NODES / 256, 256, 0, stream>>>(deg, dinv);
    agg_init_kernel <<<N_NODES, HID, 0, stream>>>(h0, dinv, b_gcn, hagg);
    agg_edges_kernel<<<N_EDGES, HID, 0, stream>>>(ei, dinv, h0, hagg);

    // 3) latent = relu(hagg) @ w_enc + b_enc   (relu fused into A staging)
    gemm_bf16_wmma<true, false, false><<<dim3(N_NODES / 128, 1), 256, 0, stream>>>(
        hagg, wenc_t, b_enc, latent, N_NODES, LAT, HID);

    // 4) tbuf = relu(latent @ w_d1 + b_d1)
    gemm_bf16_wmma<false, true, false><<<dim3(N_NODES / 128, HID / 128), 256, 0, stream>>>(
        latent, wd1_t, b_d1, tbuf, N_NODES, HID, LAT);

    // 5) recon = tbuf @ w_d2 + b_d2           (655 MB stream, NT stores)
    gemm_bf16_wmma<false, false, true><<<dim3(N_NODES / 128, (IN_DIM + 127) / 128), 256, 0, stream>>>(
        tbuf, wd2_t, b_d2, recon, N_NODES, IN_DIM, HID);
}
